// GatedAxialAttentionWidth_35029753266699
// MI455X (gfx1250) — compile-verified, matrix operand order checked by reading, not run on hardware
//
#include <hip/hip_runtime.h>
#include <hip/hip_bf16.h>
#include <math.h>

// ---------------------------------------------------------------------------
// Gated axial attention (width axis), MI455X / gfx1250, wave32 + WMMA f16.
//
// Shapes: N=16, C=256, H=128, W=128.  d = sqrt(C) = 16.
// All heavy contractions run on v_wmma_f32_16x16x32_f16 (f32 accumulate).
// Workspace layout (carved from d_ws, ~697 MB total):
//   Wq_h/Wk_h/Wv_h : 3 x 128 KB   f16  (scaled: Wq/d, Wk, Wv*Gv1)
//   rqT/rkT        : 2 x 8 MB     f16  [j][w][c]  (scaled Gq, Gk/d)
//   rv_h           : 8 MB         f16  [c][j][w]  (scaled Gv2)
//   qsT/kT         : 2 x 128 MB   f16  [ni][w][c]   (ni = n*128+i, 2048 rows)
//   v_h            : 128 MB       f16  [ni][c][w]
//   SB             : 64 MB        f16  [ni][j][w]   (Gq*qrq + Gk*krk, pre-/d)
//   PB             : 64 MB        f16  [ni][j][w]   (softmax probs)
//   O2             : 128 MB       f16  [ni][j][c]   (weight . Gv2*rv)
// ---------------------------------------------------------------------------

typedef __attribute__((ext_vector_type(16))) _Float16 v16h;
typedef __attribute__((ext_vector_type(8)))  _Float16 v8h;
typedef __attribute__((ext_vector_type(8)))  float    v8f;

#define Cc   256
#define Wd   128
#define Hh   128
#define Nn   16
#define NI   2048          // N*H rows
#define INVD 0.0625f       // 1/sqrt(256)
#define XTS  (Cc + 8)      // LDS row stride for [w][c]/[w][o] tiles

// ---- WMMA fragment helpers (wave32) ---------------------------------------
// A fragment: matrix stored row-major [M][K] (halves), row stride ld.
// B fragment: matrix stored N-major   [N][K] (halves), same load pattern.
// ISA 16-bit A 16x32 layout: lane(m=lane&15); halves0-7 = K k0+8*(lane>>4)..;
// halves8-15 = +16.
__device__ __forceinline__ v16h load_frag(const _Float16* base, int row0,
                                          int k0, int ld) {
  int lane = threadIdx.x & 31;
  int r    = lane & 15;
  int ksel = (lane >> 4) << 3;
  const _Float16* p = base + (size_t)(row0 + r) * (size_t)ld + (k0 + ksel);
  v8h lo = *(const v8h*)p;
  v8h hi = *(const v8h*)(p + 16);
  v16h out;
#pragma unroll
  for (int t = 0; t < 8; ++t) { out[t] = lo[t]; out[t + 8] = hi[t]; }
  return out;
}

__device__ __forceinline__ v8f wmma16(v16h a, v16h b, v8f c) {
  return __builtin_amdgcn_wmma_f32_16x16x32_f16(false, a, false, b,
                                                (short)0, c, false, false);
}

// D fragment: lane n = lane&15, vgpr r -> row m0 + r + 8*(lane>>4).
__device__ __forceinline__ void store_tile_f16(_Float16* base, int m0, int n0,
                                               size_t ld, v8f d) {
  int lane = threadIdx.x & 31;
  int nn = lane & 15;
  int mh = (lane >> 4) << 3;
#pragma unroll
  for (int r = 0; r < 8; ++r)
    base[(size_t)(m0 + r + mh) * ld + (n0 + nn)] = (_Float16)d[r];
}

// ---- K0: convert & pre-scale parameters -----------------------------------
__global__ __launch_bounds__(256) void gaaw_k0_prep(
    const float* __restrict__ Wq, const float* __restrict__ Wk,
    const float* __restrict__ Wv, const float* __restrict__ rq,
    const float* __restrict__ rk, const float* __restrict__ rv,
    const float* __restrict__ Gq, const float* __restrict__ Gk,
    const float* __restrict__ Gv1, const float* __restrict__ Gv2,
    _Float16* __restrict__ Wq_h, _Float16* __restrict__ Wk_h,
    _Float16* __restrict__ Wv_h, _Float16* __restrict__ rqT,
    _Float16* __restrict__ rkT, _Float16* __restrict__ rv_h) {
  const float gq = Gq[0], gk = Gk[0], gv1 = Gv1[0], gv2 = Gv2[0];
  size_t tid    = (size_t)blockIdx.x * blockDim.x + threadIdx.x;
  size_t stride = (size_t)gridDim.x * blockDim.x;
  for (size_t idx = tid; idx < (size_t)Cc * Cc; idx += stride) {
    Wq_h[idx] = (_Float16)(Wq[idx] * INVD);
    Wk_h[idx] = (_Float16)(Wk[idx]);
    Wv_h[idx] = (_Float16)(Wv[idx] * gv1);
  }
  for (size_t idx = tid; idx < (size_t)Cc * Wd * Wd; idx += stride) {
    size_t c = idx >> 14, j = (idx >> 7) & 127, w = idx & 127;
    size_t tix = (j * Wd + w) * Cc + c;            // [j][w][c]
    rqT[tix]  = (_Float16)(rq[idx] * gq);
    rkT[tix]  = (_Float16)(rk[idx] * (gk * INVD));
    rv_h[idx] = (_Float16)(rv[idx] * gv2);         // keep [c][j][w]
  }
}

// ---- K1: QKV projection per (n,i) row -------------------------------------
// Templated so every call site sees a compile-time global pointer (keeps
// address-space inference intact -> global_load_b128, not flat_load).
template <int MODE>  // 0: Q->Dst, 1: K->Dst, 2: V->global
__device__ __forceinline__ void gaaw_k1_phase(
    const _Float16* __restrict__ A, const _Float16 (*Xt)[XTS],
    _Float16 (*Dst)[XTS], _Float16* __restrict__ vout) {
  const int tid = threadIdx.x, wave = tid >> 5;
  const int lane = tid & 31, nn = lane & 15, mh = (lane >> 4) << 3;
  for (int t = wave; t < 128; t += 8) {            // 16 o-tiles x 8 w-tiles
    int m0 = (t >> 3) << 4, n0 = (t & 7) << 4;
    v16h af[8], bf[8];
#pragma unroll
    for (int kk = 0; kk < 8; ++kk) {
      af[kk] = load_frag(A, m0, kk * 32, Cc);
      bf[kk] = load_frag(&Xt[0][0], n0, kk * 32, XTS);
    }
    v8f acc = {};
#pragma unroll
    for (int kk = 0; kk < 8; ++kk) acc = wmma16(af[kk], bf[kk], acc);
    if (MODE == 2) {                               // V: direct [ni][c][w]
      store_tile_f16(vout, m0, n0, Wd, acc);
    } else {                                       // Q/K: transpose to [w][o]
      v8h h;
#pragma unroll
      for (int r = 0; r < 8; ++r) h[r] = (_Float16)acc[r];
      *(v8h*)&Dst[n0 + nn][m0 + mh] = h;           // one ds_store_b128
    }
  }
}

__global__ __launch_bounds__(256) void gaaw_k1_proj(
    const float* __restrict__ x, const _Float16* __restrict__ Wq_h,
    const _Float16* __restrict__ Wk_h, const _Float16* __restrict__ Wv_h,
    _Float16* __restrict__ qsT, _Float16* __restrict__ kT,
    _Float16* __restrict__ v_h) {
  __shared__ _Float16 Xt[Wd][XTS];     // x row, transposed [w][c]
  __shared__ _Float16 Dst[Wd][XTS];    // transpose staging [w][o]
  const int tid = threadIdx.x;
  const int ni = blockIdx.x, n = ni >> 7, i = ni & 127;
  const size_t xbase = (size_t)n * Cc * Hh * Wd + (size_t)i * Wd;

  for (int idx = tid; idx < Cc * Wd; idx += 256) {
    int c = idx >> 7, w = idx & 127;
    Xt[w][c] = (_Float16)x[xbase + (size_t)c * (Hh * Wd) + w];
  }
  __syncthreads();

  gaaw_k1_phase<0>(Wq_h, Xt, Dst, nullptr);
  __syncthreads();
  {
    _Float16* dst = qsT + (size_t)ni * Wd * Cc;
    for (int idx = tid; idx < Wd * Cc; idx += 256)
      dst[idx] = Dst[idx >> 8][idx & 255];
  }
  __syncthreads();
  gaaw_k1_phase<1>(Wk_h, Xt, Dst, nullptr);
  __syncthreads();
  {
    _Float16* dst = kT + (size_t)ni * Wd * Cc;
    for (int idx = tid; idx < Wd * Cc; idx += 256)
      dst[idx] = Dst[idx >> 8][idx & 255];
  }
  __syncthreads();
  gaaw_k1_phase<2>(Wv_h, Xt, Dst, v_h + (size_t)ni * Cc * Wd);
}

// ---- K2s: positional score terms, batched over j --------------------------
__global__ __launch_bounds__(256) void gaaw_k2s_pos(
    const _Float16* __restrict__ qsT, const _Float16* __restrict__ kT,
    const _Float16* __restrict__ rqT, const _Float16* __restrict__ rkT,
    _Float16* __restrict__ SB) {
  const int j = blockIdx.x & 127, ni0 = (blockIdx.x >> 7) << 7;
  const int wave = threadIdx.x >> 5;
  const _Float16* Aq = qsT + (size_t)j * Cc;       // rows ni, ld W*C
  const _Float16* Ak = kT + (size_t)j * Cc;
  const _Float16* Bq = rqT + (size_t)j * Wd * Cc;  // rows w, ld C
  const _Float16* Bk = rkT + (size_t)j * Wd * Cc;
  for (int t = wave; t < 64; t += 8) {             // 8 ni-tiles x 8 w-tiles
    int m0 = ni0 + ((t >> 3) << 4), n0 = (t & 7) << 4;
    v8f acc = {};
    v16h af[8], bf[8];
#pragma unroll
    for (int kk = 0; kk < 8; ++kk) {
      af[kk] = load_frag(Aq, m0, kk * 32, Wd * Cc);
      bf[kk] = load_frag(Bq, n0, kk * 32, Cc);
    }
#pragma unroll
    for (int kk = 0; kk < 8; ++kk) acc = wmma16(af[kk], bf[kk], acc);
#pragma unroll
    for (int kk = 0; kk < 8; ++kk) {
      af[kk] = load_frag(Ak, m0, kk * 32, Wd * Cc);
      bf[kk] = load_frag(Bk, n0, kk * 32, Cc);
    }
#pragma unroll
    for (int kk = 0; kk < 8; ++kk) acc = wmma16(af[kk], bf[kk], acc);
    store_tile_f16(SB + (size_t)j * Wd, m0, n0, Wd * Wd, acc);
  }
}

// ---- K2: qk scores + softmax per (n,i) row --------------------------------
__global__ __launch_bounds__(256) void gaaw_k2_softmax(
    const _Float16* __restrict__ qsT, const _Float16* __restrict__ kT,
    const _Float16* __restrict__ SB, _Float16* __restrict__ PB) {
  __shared__ float S[Wd][Wd + 4];
  const int tid = threadIdx.x, wave = tid >> 5, ni = blockIdx.x;
  const _Float16* A = qsT + (size_t)ni * Wd * Cc;  // rows j, ld C
  const _Float16* B = kT + (size_t)ni * Wd * Cc;   // rows w, ld C
  const _Float16* SBr = SB + (size_t)ni * Wd * Wd;

  for (int t = wave; t < 64; t += 8) {
    int m0 = (t >> 3) << 4, n0 = (t & 7) << 4;
    v16h af[8], bf[8];
#pragma unroll
    for (int kk = 0; kk < 8; ++kk) {
      af[kk] = load_frag(A, m0, kk * 32, Cc);
      bf[kk] = load_frag(B, n0, kk * 32, Cc);
    }
    v8f acc = {};
#pragma unroll
    for (int kk = 0; kk < 8; ++kk) acc = wmma16(af[kk], bf[kk], acc);
    int lane = tid & 31, nn = lane & 15, mh = (lane >> 4) << 3;
#pragma unroll
    for (int r = 0; r < 8; ++r) {
      int jj = m0 + r + mh, ww = n0 + nn;
      S[jj][ww] = acc[r] + (float)SBr[(size_t)jj * Wd + ww];
    }
  }
  __syncthreads();
  if (tid < Wd) {
    int j = tid;
    float mx = -3.4e38f;
    for (int w = 0; w < Wd; ++w) mx = fmaxf(mx, S[j][w]);
    float sum = 0.f;
    for (int w = 0; w < Wd; ++w) {
      float e = __expf(S[j][w] - mx);
      S[j][w] = e; sum += e;
    }
    float inv = 1.f / sum;
    for (int w = 0; w < Wd; ++w) S[j][w] *= inv;
  }
  __syncthreads();
  for (int idx = tid; idx < Wd * Wd; idx += 256)
    PB[(size_t)ni * Wd * Wd + idx] = (_Float16)S[idx >> 7][idx & 127];
}

// ---- K3: O2[ni][j][c] = P_j . (Gv2*rv)_j, batched over j ------------------
__global__ __launch_bounds__(256) void gaaw_k3_out2(
    const _Float16* __restrict__ PB, const _Float16* __restrict__ rv_h,
    _Float16* __restrict__ O2) {
  const int j = blockIdx.x & 127, ni0 = (blockIdx.x >> 7) << 7;
  const int wave = threadIdx.x >> 5;
  const _Float16* A = PB + (size_t)j * Wd;     // rows ni, ld W*W, K=w
  const _Float16* B = rv_h + (size_t)j * Wd;   // rows c,  ld W*W, K=w
  for (int t = wave; t < 128; t += 8) {        // 8 ni-tiles x 16 c-tiles
    int m0 = ni0 + ((t >> 4) << 4), n0 = (t & 15) << 4;
    v16h af[4], bf[4];
#pragma unroll
    for (int kk = 0; kk < 4; ++kk) {
      af[kk] = load_frag(A, m0, kk * 32, Wd * Wd);
      bf[kk] = load_frag(B, n0, kk * 32, Wd * Wd);
    }
    v8f acc = {};
#pragma unroll
    for (int kk = 0; kk < 4; ++kk) acc = wmma16(af[kk], bf[kk], acc);
    store_tile_f16(O2 + (size_t)j * Cc, m0, n0, Wd * Cc, acc);
  }
}

// ---- K4: out = Gv1*v . P^T + O2^T, per (n,i) row, NCHW coalesced ----------
__global__ __launch_bounds__(256) void gaaw_k4_final(
    const _Float16* __restrict__ v_h, const _Float16* __restrict__ PB,
    const _Float16* __restrict__ O2, float* __restrict__ out) {
  __shared__ _Float16 P[Wd][136];          // [j][w], 272B rows (16B aligned)
  __shared__ _Float16 O2s[Wd][XTS];        // [j][c]
  const int tid = threadIdx.x, wave = tid >> 5;
  const int ni = blockIdx.x, n = ni >> 7, i = ni & 127;

  for (int idx = tid; idx < Wd * Wd; idx += 256)
    P[idx >> 7][idx & 127] = PB[(size_t)ni * Wd * Wd + idx];
  for (int idx = tid; idx < Wd * Cc; idx += 256)
    O2s[idx >> 8][idx & 255] = O2[(size_t)ni * Wd * Cc + idx];
  __syncthreads();

  const _Float16* A = v_h + (size_t)ni * Cc * Wd;  // rows c, ld W, K=w
  float* outb = out + (size_t)n * Cc * Hh * Wd + (size_t)i * Wd;
  const int lane = tid & 31, nn = lane & 15, mh = (lane >> 4) << 3;
  for (int t = wave; t < 128; t += 8) {            // 16 c-tiles x 8 j-tiles
    int m0 = (t >> 3) << 4, n0 = (t & 7) << 4;
    v16h af[4], bf[4];
#pragma unroll
    for (int kk = 0; kk < 4; ++kk) {
      af[kk] = load_frag(A, m0, kk * 32, Wd);
      bf[kk] = load_frag(&P[0][0], n0, kk * 32, 136);
    }
    v8f acc = {};
#pragma unroll
    for (int kk = 0; kk < 4; ++kk) acc = wmma16(af[kk], bf[kk], acc);
#pragma unroll
    for (int r = 0; r < 8; ++r) {
      int c = m0 + r + mh, jj = n0 + nn;
      outb[(size_t)c * (Hh * Wd) + jj] = acc[r] + (float)O2s[jj][c];
    }
  }
}

// ---------------------------------------------------------------------------
extern "C" void kernel_launch(void* const* d_in, const int* in_sizes, int n_in,
                              void* d_out, int out_size, void* d_ws,
                              size_t ws_size, hipStream_t stream) {
  const float* x   = (const float*)d_in[0];
  const float* Wq  = (const float*)d_in[1];
  const float* Wk  = (const float*)d_in[2];
  const float* Wv  = (const float*)d_in[3];
  const float* rq  = (const float*)d_in[4];
  const float* rk  = (const float*)d_in[5];
  const float* rv  = (const float*)d_in[6];
  const float* Gq  = (const float*)d_in[7];
  const float* Gk  = (const float*)d_in[8];
  const float* Gv1 = (const float*)d_in[9];
  const float* Gv2 = (const float*)d_in[10];
  float* out = (float*)d_out;

  char* ws = (char*)d_ws;
  size_t off = 0;
  auto carve = [&](size_t bytes) {
    void* p = ws + off;
    off += (bytes + 255) & ~(size_t)255;
    return p;
  };
  _Float16* Wq_h = (_Float16*)carve((size_t)Cc * Cc * 2);
  _Float16* Wk_h = (_Float16*)carve((size_t)Cc * Cc * 2);
  _Float16* Wv_h = (_Float16*)carve((size_t)Cc * Cc * 2);
  _Float16* rqT  = (_Float16*)carve((size_t)Cc * Wd * Wd * 2);
  _Float16* rkT  = (_Float16*)carve((size_t)Cc * Wd * Wd * 2);
  _Float16* rv_h = (_Float16*)carve((size_t)Cc * Wd * Wd * 2);
  _Float16* qsT  = (_Float16*)carve((size_t)NI * Wd * Cc * 2);
  _Float16* kT   = (_Float16*)carve((size_t)NI * Wd * Cc * 2);
  _Float16* v_h  = (_Float16*)carve((size_t)NI * Cc * Wd * 2);
  _Float16* SB   = (_Float16*)carve((size_t)NI * Wd * Wd * 2);
  _Float16* PB   = (_Float16*)carve((size_t)NI * Wd * Wd * 2);
  _Float16* O2   = (_Float16*)carve((size_t)NI * Wd * Cc * 2);
  // total carved ~697 MB; assumes ws_size covers it.

  gaaw_k0_prep<<<2048, 256, 0, stream>>>(Wq, Wk, Wv, rq, rk, rv, Gq, Gk, Gv1,
                                         Gv2, Wq_h, Wk_h, Wv_h, rqT, rkT, rv_h);
  gaaw_k1_proj<<<NI, 256, 0, stream>>>(x, Wq_h, Wk_h, Wv_h, qsT, kT, v_h);
  gaaw_k2s_pos<<<NI, 256, 0, stream>>>(qsT, kT, rqT, rkT, SB);
  gaaw_k2_softmax<<<NI, 256, 0, stream>>>(qsT, kT, SB, PB);
  gaaw_k3_out2<<<NI, 256, 0, stream>>>(PB, rv_h, O2);
  gaaw_k4_final<<<NI, 256, 0, stream>>>(v_h, PB, O2, out);
}
